// AFNO2D_6176162971872
// MI455X (gfx1250) — matrix operand order repaired and checked by reading, not compiled, not run on hardware
//
#include <hip/hip_runtime.h>
#include <hip/hip_bf16.h>

// AFNO2D on gfx1250: every stage (rfft-W, fft-H, block-diag complex MLP,
// ifft-H, irfft-W + residual) is a batched GEMM on v_wmma_f32_16x16x32_f16.
// f32 accumulate, f16 operands/intermediates (planar re/im), DFT matrices
// built on device once per launch. MLP weights are staged into LDS with the
// Tensor Data Mover (tensor_load_to_lds + s_wait_tensorcnt).

typedef __attribute__((ext_vector_type(16))) _Float16 v16h;
typedef __attribute__((ext_vector_type(8)))  float    v8f;
typedef __attribute__((ext_vector_type(4)))  unsigned int u32x4;
typedef __attribute__((ext_vector_type(4)))  int      i32x4;
typedef __attribute__((ext_vector_type(8)))  int      i32x8;

#define DEVI __device__ __forceinline__

namespace afno {
constexpr int Bsz = 4;
constexpr int Hs  = 128;
constexpr int Wsp = 128;
constexpr int Cs  = 768;
constexpr int Wc  = 65;
constexpr int Wcp = 80;             // Wc padded to a multiple of 16
constexpr int NB  = 8;
constexpr int BK  = 96;             // channels per block
constexpr int WMAT = NB * BK * BK;  // 73728 elements per real weight matrix
}
using namespace afno;

DEVI v8f zero8() {
  v8f z;
#pragma unroll
  for (int i = 0; i < 8; ++i) z[i] = 0.f;
  return z;
}

DEVI v8f wmma_f16(v16h a, v16h b, v8f c) {
  // D = A(16x32 f16) * B(32x16 f16) + C(16x16 f32)
  return __builtin_amdgcn_wmma_f32_16x16x32_f16(false, a, false, b, (short)0, c,
                                                false, false);
}

// ---- fragment builders per CDNA5 ISA 7.12.2 (wave32) --------------------
// A (16x32, MxK): lanes 0-15 -> M=lane, lanes 16-31 -> M=lane-16.
// VGPR v<4 packs K = 8*half + 2v, +1 ; v>=4 packs K = 16 + 8*half + 2(v-4), +1.
template <typename F>
DEVI v16h frag_A(F elem) {
  const int lane = threadIdx.x & 31;
  const int half = lane >> 4, m = lane & 15;
  v16h a;
#pragma unroll
  for (int v = 0; v < 8; ++v) {
    const int kb = (v < 4) ? (8 * half + 2 * v) : (16 + 8 * half + 2 * (v - 4));
    a[2 * v]     = elem(m, kb);
    a[2 * v + 1] = elem(m, kb + 1);
  }
  return a;
}

// B (32x16, KxN): lane -> N = lane&15 ; VGPR v<4 packs K = 8*half + 2v, +1 ; v>=4 packs K = 16 + 8*half + 2(v-4), +1.
template <typename F>
DEVI v16h frag_B(F elem) {
  const int lane = threadIdx.x & 31;
  const int half = lane >> 4, n = lane & 15;
  v16h b;
#pragma unroll
  for (int v = 0; v < 8; ++v) {
    const int k = (v < 4) ? (8 * half + 2 * v) : (16 + 8 * half + 2 * (v - 4));
    b[2 * v]     = elem(k, n);
    b[2 * v + 1] = elem(k + 1, n);
  }
  return b;
}

// D (16x16 f32): lane -> N = lane&15 ; VGPR v -> M = v + 8*(lane>>4).
template <typename F>
DEVI void frag_D(const v8f& d, F put) {
  const int lane = threadIdx.x & 31;
  const int half = lane >> 4, n = lane & 15;
#pragma unroll
  for (int v = 0; v < 8; ++v) put(v + 8 * half, n, d[v]);
}

// ---- Tensor Data Mover: 2-D f16 tile (dim0 x dim1, row stride stride0) ---
// D# per CDNA5 ISA ch.8: group0 = {count|flags, lds_addr, gaddr_lo,
// gaddr_hi|type=2}; group1 = dims/strides, data_size=2B; groups 2/3 zero.
DEVI void tdm_load_2d_f16(unsigned int lds_off, const void* gptr, int dim0,
                          int dim1, int stride0) {
  const unsigned long long ga = (unsigned long long)(uintptr_t)gptr;
  u32x4 g0;
  g0[0] = 1u;                                        // count=1, no gather
  g0[1] = lds_off;                                   // lds_addr (bytes)
  g0[2] = (unsigned int)(ga & 0xFFFFFFFFull);        // global_addr[31:0]
  g0[3] = (unsigned int)((ga >> 32) & 0x01FFFFFFull) // global_addr[56:32]
          | (2u << 30);                              // type = 2 ("image")
  i32x8 g1;
  g1[0] = 0x00010000;                 // data_size=1 (2 bytes), mask/flags=0
  g1[1] = (dim0 & 0xFFFF) << 16;      // tensor_dim0[15:0]
  g1[2] = (dim1 & 0xFFFF) << 16;      // tensor_dim1[15:0] (dim0 hi = 0)
  g1[3] = (dim0 & 0xFFFF) << 16;      // tile_dim0        (dim1 hi = 0)
  g1[4] = (dim1 & 0xFFFF);            // tile_dim1 ; tile_dim2 = 0
  g1[5] = stride0;                    // tensor_dim0_stride[31:0]
  g1[6] = 0;
  g1[7] = 0;
  i32x4 z4;
#pragma unroll
  for (int i = 0; i < 4; ++i) z4[i] = 0;
#if defined(__clang_major__) && (__clang_major__ >= 23)
  i32x8 z8;
#pragma unroll
  for (int i = 0; i < 8; ++i) z8[i] = 0;
  __builtin_amdgcn_tensor_load_to_lds(g0, g1, z4, z4, z8, 0);
#else
  __builtin_amdgcn_tensor_load_to_lds(g0, g1, z4, z4, 0);
#endif
}

// ===================== setup kernels (tiny, once per launch) ==============

__global__ void k_build_F(_Float16* __restrict__ Fr, _Float16* __restrict__ Fi) {
  // forward rfft along W: F[wc][w] = exp(-2pi i wc*w/128)/sqrt(128); rows >=65 zero
  const int idx = blockIdx.x * 256 + threadIdx.x;   // 80*128 = 10240
  if (idx >= Wcp * 128) return;
  const int wc = idx / 128, w = idx % 128;
  float vr = 0.f, vi = 0.f;
  if (wc < Wc) {
    const int p = (wc * w) & 127;
    const float a = 0.049087385212340517f * (float)p;  // 2pi/128 * p
    const float s = 0.08838834764831845f;              // 1/sqrt(128)
    vr = cosf(a) * s;
    vi = -sinf(a) * s;
  }
  Fr[idx] = (_Float16)vr;
  Fi[idx] = (_Float16)vi;
}

__global__ void k_build_cmat(_Float16* __restrict__ Mr, _Float16* __restrict__ Mi,
                             float sign) {
  // complex DFT along H: M[i][j] = exp(sign * 2pi i*j/128)/sqrt(128)  (symmetric)
  const int idx = blockIdx.x * 256 + threadIdx.x;   // 128*128 = 16384
  if (idx >= 128 * 128) return;
  const int i = idx / 128, j = idx % 128;
  const int p = (i * j) & 127;
  const float a = 0.049087385212340517f * (float)p;
  const float s = 0.08838834764831845f;
  Mr[idx] = (_Float16)(cosf(a) * s);
  Mi[idx] = (_Float16)(sign * sinf(a) * s);
}

__global__ void k_build_FI(_Float16* __restrict__ FI) {
  // irfft along W as one real matrix [128][192]:
  //   cols   0.. 79 : ar[w][wc] =  c_wc * cos(2pi wc w/128)/sqrt(128)
  //   cols  96..175 : ai[w][wc] = -c_wc * sin(2pi wc w/128)/sqrt(128)
  //   c_0 = c_64 = 1, else 2 ; all other cols zero (K-tile padding)
  const int idx = blockIdx.x * 256 + threadIdx.x;   // 128*192 = 24576
  if (idx >= 128 * 192) return;
  const int w = idx / 192, k = idx % 192;
  float v = 0.f;
  const float s = 0.08838834764831845f;
  if (k < Wcp) {
    const int wc = k;
    if (wc < Wc) {
      const float c = (wc == 0 || wc == 64) ? 1.f : 2.f;
      const int p = (wc * w) & 127;
      v = c * cosf(0.049087385212340517f * (float)p) * s;
    }
  } else if (k >= 96 && k < 96 + Wcp) {
    const int wc = k - 96;
    if (wc < Wc) {
      const float c = (wc == 0 || wc == 64) ? 1.f : 2.f;
      const int p = (wc * w) & 127;
      v = -c * sinf(0.049087385212340517f * (float)p) * s;
    }
  }
  FI[idx] = (_Float16)v;
}

__global__ void k_conv_w(const float* __restrict__ w1, const float* __restrict__ w2,
                         _Float16* __restrict__ W1r, _Float16* __restrict__ W1i,
                         _Float16* __restrict__ W2r, _Float16* __restrict__ W2i) {
  const int idx = blockIdx.x * 256 + threadIdx.x;   // 73728 exact
  if (idx >= WMAT) return;
  W1r[idx] = (_Float16)w1[idx];
  W1i[idx] = (_Float16)w1[WMAT + idx];
  W2r[idx] = (_Float16)w2[idx];
  W2i[idx] = (_Float16)w2[WMAT + idx];
}

// ===================== stage 1: rfft along W (real GEMM x2) ===============
// out[b,h,wc,c] (re,im) = F[wc,:] . x[b,h,:,c]
__global__ __launch_bounds__(128) void k_stage1(
    const float* __restrict__ x, const _Float16* __restrict__ Fr,
    const _Float16* __restrict__ Fi, _Float16* __restrict__ outR,
    _Float16* __restrict__ outI) {
  const int waveId = threadIdx.x >> 5;
  const int cbase  = blockIdx.x * 64 + waveId * 16;
  const int wcbase = blockIdx.y * 16;
  const int bh     = blockIdx.z;                     // b*H + h
  const float* xrow = x + (size_t)bh * Wsp * Cs;
  v8f accR = zero8(), accI = zero8();
#pragma unroll
  for (int kt = 0; kt < 4; ++kt) {
    const _Float16* fr = Fr + wcbase * 128 + kt * 32;
    const _Float16* fi = Fi + wcbase * 128 + kt * 32;
    v16h aR = frag_A([&](int m, int k) -> _Float16 { return fr[m * 128 + k]; });
    v16h aI = frag_A([&](int m, int k) -> _Float16 { return fi[m * 128 + k]; });
    const float* xb = xrow + (size_t)(kt * 32) * Cs + cbase;
    v16h bX = frag_B([&](int k, int n) -> _Float16 {
      return (_Float16)xb[(size_t)k * Cs + n];
    });
    accR = wmma_f16(aR, bX, accR);
    accI = wmma_f16(aI, bX, accI);
  }
  _Float16* oR = outR + ((size_t)bh * Wcp + wcbase) * Cs + cbase;
  _Float16* oI = outI + ((size_t)bh * Wcp + wcbase) * Cs + cbase;
  frag_D(accR, [&](int m, int n, float v) { oR[(size_t)m * Cs + n] = (_Float16)v; });
  frag_D(accI, [&](int m, int n, float v) { oI[(size_t)m * Cs + n] = (_Float16)v; });
}

// ===================== stages 2 & 4: complex GEMM along H =================
// out[b,hk,wc,c] = sum_h M[hk,h] * in[b,h,wc,c]   (complex * complex)
// Re part via two accumulators (Mr.Yr and Mi.Yi), subtracted at store.
__global__ __launch_bounds__(128) void k_fft_h(
    const _Float16* __restrict__ inR, const _Float16* __restrict__ inI,
    const _Float16* __restrict__ Mr, const _Float16* __restrict__ Mi,
    _Float16* __restrict__ outR, _Float16* __restrict__ outI) {
  const int waveId = threadIdx.x >> 5;
  const int cbase  = blockIdx.x * 64 + waveId * 16;
  const int hkbase = blockIdx.y * 16;
  const int bwc    = blockIdx.z;                     // b*Wcp + wc
  const int b = bwc / Wcp, wc = bwc % Wcp;
  const size_t hStride = (size_t)Wcp * Cs;
  const size_t colOff  = (size_t)b * Hs * hStride + (size_t)wc * Cs + cbase;
  v8f accRp = zero8(), accRm = zero8(), accI = zero8();
#pragma unroll
  for (int kt = 0; kt < 4; ++kt) {
    const _Float16* mr = Mr + hkbase * 128 + kt * 32;
    const _Float16* mi = Mi + hkbase * 128 + kt * 32;
    v16h aR = frag_A([&](int m, int k) -> _Float16 { return mr[m * 128 + k]; });
    v16h aI = frag_A([&](int m, int k) -> _Float16 { return mi[m * 128 + k]; });
    const _Float16* ir = inR + colOff + (size_t)(kt * 32) * hStride;
    const _Float16* ii = inI + colOff + (size_t)(kt * 32) * hStride;
    v16h bR = frag_B([&](int k, int n) -> _Float16 { return ir[(size_t)k * hStride + n]; });
    v16h bI = frag_B([&](int k, int n) -> _Float16 { return ii[(size_t)k * hStride + n]; });
    accRp = wmma_f16(aR, bR, accRp);
    accRm = wmma_f16(aI, bI, accRm);
    accI  = wmma_f16(aR, bI, accI);
    accI  = wmma_f16(aI, bR, accI);
  }
  v8f accR;
#pragma unroll
  for (int i = 0; i < 8; ++i) accR[i] = accRp[i] - accRm[i];
  _Float16* oR = outR + ((size_t)(b * Hs + hkbase) * Wcp + wc) * Cs + cbase;
  _Float16* oI = outI + ((size_t)(b * Hs + hkbase) * Wcp + wc) * Cs + cbase;
  frag_D(accR, [&](int m, int n, float v) { oR[(size_t)m * hStride + n] = (_Float16)v; });
  frag_D(accI, [&](int m, int n, float v) { oI[(size_t)m * hStride + n] = (_Float16)v; });
}

// ===================== stage 3: block MLP + soft-shrink ===================
// Weights staged into LDS by the Tensor Data Mover; LDS buffer reused for
// layer 2 after the transpose barrier.
__global__ __launch_bounds__(128) void k_mlp(
    const _Float16* __restrict__ inR, const _Float16* __restrict__ inI,
    const _Float16* __restrict__ W1r, const _Float16* __restrict__ W1i,
    const _Float16* __restrict__ W2r, const _Float16* __restrict__ W2i,
    const float* __restrict__ b1, const float* __restrict__ b2,
    _Float16* __restrict__ outR, _Float16* __restrict__ outI) {
  const int blk    = blockIdx.x;                  // channel block 0..7
  const int waveId = threadIdx.x >> 5;
  const int rt     = blockIdx.y * 4 + waveId;     // row tile (16 positions)
  const int r0     = rt * 16;
  __shared__ _Float16 wr_lds[BK * BK];            // 18432 B (W?r tile)
  __shared__ _Float16 wi_lds[BK * BK];            // 18432 B (W?i tile)
  __shared__ _Float16 l1[4][2][16][98];           // 25088 B transpose buffer
  _Float16 (*lr)[98] = l1[waveId][0];
  _Float16 (*li)[98] = l1[waveId][1];

  const float* b1r = b1 + blk * BK;
  const float* b1i = b1 + NB * BK + blk * BK;
  const float* b2r = b2 + blk * BK;
  const float* b2i = b2 + NB * BK + blk * BK;

  const int lane = threadIdx.x & 31;
  const int halfSel = lane >> 4, n = lane & 15;

  // ---- TDM: stage layer-1 weights (96x96 f16 each) into LDS ----
  if (waveId == 0) {
    tdm_load_2d_f16((unsigned int)(uintptr_t)&wr_lds[0], W1r + blk * BK * BK,
                    BK, BK, BK);
    tdm_load_2d_f16((unsigned int)(uintptr_t)&wi_lds[0], W1i + blk * BK * BK,
                    BK, BK, BK);
    __builtin_amdgcn_s_wait_tensorcnt(0);
  }

  // ---- layer 1 A fragments from global (16 rows x 96 K), cached ----
  const _Float16* xr = inR + (size_t)r0 * Cs + blk * BK;
  const _Float16* xi = inI + (size_t)r0 * Cs + blk * BK;
  v16h aXr[3], aXi[3];
#pragma unroll
  for (int kt = 0; kt < 3; ++kt) {
    aXr[kt] = frag_A([&](int m, int k) -> _Float16 { return xr[(size_t)m * Cs + kt * 32 + k]; });
    aXi[kt] = frag_A([&](int m, int k) -> _Float16 { return xi[(size_t)m * Cs + kt * 32 + k]; });
  }
  __syncthreads();   // publish W1 tiles

#pragma unroll
  for (int ct = 0; ct < 6; ++ct) {
    v8f arp = zero8(), arm = zero8(), ai = zero8();
#pragma unroll
    for (int kt = 0; kt < 3; ++kt) {
      v16h bWr = frag_B([&](int k, int nn) -> _Float16 {
        return wr_lds[(kt * 32 + k) * BK + ct * 16 + nn];
      });
      v16h bWi = frag_B([&](int k, int nn) -> _Float16 {
        return wi_lds[(kt * 32 + k) * BK + ct * 16 + nn];
      });
      arp = wmma_f16(aXr[kt], bWr, arp);
      arm = wmma_f16(aXi[kt], bWi, arm);
      ai  = wmma_f16(aXi[kt], bWr, ai);
      ai  = wmma_f16(aXr[kt], bWi, ai);
    }
    const float br = b1r[ct * 16 + n], bi = b1i[ct * 16 + n];
#pragma unroll
    for (int v = 0; v < 8; ++v) {
      const int m = v + 8 * halfSel;
      float vr = arp[v] - arm[v] + br; vr = vr > 0.f ? vr : 0.f;
      float vi = ai[v] + bi;           vi = vi > 0.f ? vi : 0.f;
      lr[m][ct * 16 + n] = (_Float16)vr;
      li[m][ct * 16 + n] = (_Float16)vi;
    }
  }
  __syncthreads();   // D-layout -> A-layout transpose; W1 tiles now dead

  // ---- TDM: overwrite LDS weight buffer with layer-2 weights ----
  if (waveId == 0) {
    tdm_load_2d_f16((unsigned int)(uintptr_t)&wr_lds[0], W2r + blk * BK * BK,
                    BK, BK, BK);
    tdm_load_2d_f16((unsigned int)(uintptr_t)&wi_lds[0], W2i + blk * BK * BK,
                    BK, BK, BK);
    __builtin_amdgcn_s_wait_tensorcnt(0);
  }

  // ---- layer 2 A fragments from LDS ----
  v16h aOr[3], aOi[3];
#pragma unroll
  for (int kt = 0; kt < 3; ++kt) {
    aOr[kt] = frag_A([&](int m, int k) -> _Float16 { return lr[m][kt * 32 + k]; });
    aOi[kt] = frag_A([&](int m, int k) -> _Float16 { return li[m][kt * 32 + k]; });
  }
  __syncthreads();   // publish W2 tiles

#pragma unroll
  for (int ct = 0; ct < 6; ++ct) {
    v8f arp = zero8(), arm = zero8(), ai = zero8();
#pragma unroll
    for (int kt = 0; kt < 3; ++kt) {
      v16h bWr = frag_B([&](int k, int nn) -> _Float16 {
        return wr_lds[(kt * 32 + k) * BK + ct * 16 + nn];
      });
      v16h bWi = frag_B([&](int k, int nn) -> _Float16 {
        return wi_lds[(kt * 32 + k) * BK + ct * 16 + nn];
      });
      arp = wmma_f16(aOr[kt], bWr, arp);
      arm = wmma_f16(aOi[kt], bWi, arm);
      ai  = wmma_f16(aOi[kt], bWr, ai);
      ai  = wmma_f16(aOr[kt], bWi, ai);
    }
    const float cbr = b2r[ct * 16 + n], cbi = b2i[ct * 16 + n];
#pragma unroll
    for (int v = 0; v < 8; ++v) {
      const int m = v + 8 * halfSel;
      const int r = r0 + m;
      const int wc = r % Wcp;
      const int hk = (r / Wcp) % Hs;
      const int kh = (hk < 64) ? hk : hk - 128;
      const float k2  = (float)(kh * kh + wc * wc);
      const float thr = fmaxf(0.01f * expf(-0.5f * k2 * (1.f / 400.f)), 0.0005f);
      const float orv = arp[v] - arm[v] + cbr;
      const float oiv = ai[v] + cbi;
      const float mag = sqrtf(orv * orv + oiv * oiv + 1e-8f);
      const float sf  = fmaxf(mag - thr, 0.f) / mag;
      const size_t o = (size_t)r * Cs + blk * BK + ct * 16 + n;
      outR[o] = (_Float16)(orv * sf);
      outI[o] = (_Float16)(oiv * sf);
    }
  }
}

// ===================== stage 5: irfft along W + residual ==================
// y[b,h,w,c] = FI[w,:] . [Re;Im][b,h,:,c] + x[b,h,w,c]
__global__ __launch_bounds__(128) void k_stage5(
    const _Float16* __restrict__ inR, const _Float16* __restrict__ inI,
    const _Float16* __restrict__ FI, const float* __restrict__ x,
    float* __restrict__ out) {
  const int waveId = threadIdx.x >> 5;
  const int cbase  = blockIdx.x * 64 + waveId * 16;
  const int wbase  = blockIdx.y * 16;
  const int bh     = blockIdx.z;
  const _Float16* pR = inR + (size_t)bh * Wcp * Cs + cbase;
  const _Float16* pI = inI + (size_t)bh * Wcp * Cs + cbase;
  v8f acc = zero8();
#pragma unroll
  for (int kt = 0; kt < 6; ++kt) {
    const _Float16* fi = FI + wbase * 192 + kt * 32;
    v16h aF = frag_A([&](int m, int k) -> _Float16 { return fi[m * 192 + k]; });
    v16h bS = frag_B([&](int k, int n) -> _Float16 {
      const int kk = kt * 32 + k;
      if (kk < Wcp) return pR[(size_t)kk * Cs + n];
      if (kk >= 96 && kk < 96 + Wcp) return pI[(size_t)(kk - 96) * Cs + n];
      return (_Float16)0.f;
    });
    acc = wmma_f16(aF, bS, acc);
  }
  const float* xr = x + ((size_t)bh * Wsp + wbase) * Cs + cbase;
  float* o = out + ((size_t)bh * Wsp + wbase) * Cs + cbase;
  frag_D(acc, [&](int m, int n, float v) {
    o[(size_t)m * Cs + n] = v + xr[(size_t)m * Cs + n];
  });
}

// ===================== host launcher ======================================
extern "C" void kernel_launch(void* const* d_in, const int* in_sizes, int n_in,
                              void* d_out, int out_size, void* d_ws, size_t ws_size,
                              hipStream_t stream) {
  (void)in_sizes; (void)n_in; (void)out_size;
  const float* x  = (const float*)d_in[0];
  const float* w1 = (const float*)d_in[1];
  const float* b1 = (const float*)d_in[2];
  const float* w2 = (const float*)d_in[3];
  const float* b2 = (const float*)d_in[4];
  float* out = (float*)d_out;

  _Float16* wsp = (_Float16*)d_ws;
  const size_t P = (size_t)Bsz * Hs * Wcp * Cs;   // 31,457,280 elements / plane
  size_t off = 0;
  _Float16* Are = wsp + off; off += P;
  _Float16* Aim = wsp + off; off += P;
  _Float16* Bre = wsp + off; off += P;
  _Float16* Bim = wsp + off; off += P;
  _Float16* Fr  = wsp + off; off += Wcp * 128;
  _Float16* Fi  = wsp + off; off += Wcp * 128;
  _Float16* Gr  = wsp + off; off += 128 * 128;
  _Float16* Gi  = wsp + off; off += 128 * 128;
  _Float16* GIr = wsp + off; off += 128 * 128;
  _Float16* GIi = wsp + off; off += 128 * 128;
  _Float16* FIm = wsp + off; off += 128 * 192;
  _Float16* W1r = wsp + off; off += WMAT;
  _Float16* W1i = wsp + off; off += WMAT;
  _Float16* W2r = wsp + off; off += WMAT;
  _Float16* W2i = wsp + off; off += WMAT;
  if (ws_size < off * sizeof(_Float16)) return;   // need ~241 MB scratch

  // setup: DFT matrices + f16 weights (cheap, once per launch, graph-safe)
  k_build_F   <<<dim3(40), dim3(256), 0, stream>>>(Fr, Fi);
  k_build_cmat<<<dim3(64), dim3(256), 0, stream>>>(Gr, Gi, -1.f);   // forward
  k_build_cmat<<<dim3(64), dim3(256), 0, stream>>>(GIr, GIi, 1.f);  // inverse
  k_build_FI  <<<dim3(96), dim3(256), 0, stream>>>(FIm);
  k_conv_w    <<<dim3(288), dim3(256), 0, stream>>>(w1, w2, W1r, W1i, W2r, W2i);

  // stage 1: rfft-W       x (f32) -> A (f16 re/im)
  k_stage1<<<dim3(Cs / 64, Wcp / 16, Bsz * Hs), dim3(128), 0, stream>>>(
      x, Fr, Fi, Are, Aim);
  // stage 2: fft-H        A -> B
  k_fft_h<<<dim3(Cs / 64, Hs / 16, Bsz * Wcp), dim3(128), 0, stream>>>(
      Are, Aim, Gr, Gi, Bre, Bim);
  // stage 3: block MLP + soft-shrink   B -> A
  k_mlp<<<dim3(NB, (Bsz * Hs * Wcp / 16) / 4), dim3(128), 0, stream>>>(
      Bre, Bim, W1r, W1i, W2r, W2i, b1, b2, Are, Aim);
  // stage 4: ifft-H       A -> B
  k_fft_h<<<dim3(Cs / 64, Hs / 16, Bsz * Wcp), dim3(128), 0, stream>>>(
      Are, Aim, GIr, GIi, Bre, Bim);
  // stage 5: irfft-W + residual   B (+x) -> out
  k_stage5<<<dim3(Cs / 64, Wsp / 16, Bsz * Hs), dim3(128), 0, stream>>>(
      Bre, Bim, FIm, x, out);
}
